// MomentumSSM_82961588290167
// MI455X (gfx1250) — matrix operand order repaired and checked by reading, not compiled, not run on hardware
//
#include <hip/hip_runtime.h>
#include <math.h>

// ---------------------------------------------------------------------------
// MomentumSSM for MI455X (gfx1250, wave32).
//   D_INNER=512, D_STATE=16, DT_RANK=32, L=4096, CHUNK=32, NC=128
// K1: WMMA projections (x@W_xp^T, softplus(dt_in@W_dt^T + b_dt))
// K2: chunk-local scan stats (zero ICs), async global->LDS tile staging
// K3: sequential carry chain over chunks (with global_prefetch)
// K4: exact in-chunk recurrence w/ carries, shfl reduce over n, write y
// ---------------------------------------------------------------------------

#define D_INNER 512
#define D_STATE 16
#define DT_RANK 32
#define L_SEQ   4096
#define CS      32
#define NC      (L_SEQ / CS)   // 128

typedef __attribute__((ext_vector_type(16))) _Float16 v16h;
typedef __attribute__((ext_vector_type(8)))  float    v8f;
typedef __attribute__((ext_vector_type(4)))  int      v4i;

#if defined(__AMDGCN__) && __has_builtin(__builtin_amdgcn_global_load_async_to_lds_b128)
#define HAS_ASYNC_LDS 1
typedef __attribute__((address_space(1))) v4i* gas_v4ip;   // global (as1)
typedef __attribute__((address_space(3))) v4i* las_v4ip;   // LDS (as3)
#else
#define HAS_ASYNC_LDS 0
#endif

__device__ __forceinline__ float sigmoidf_(float z) { return 1.0f / (1.0f + __expf(-z)); }
__device__ __forceinline__ float softplusf_(float z) {
    return (z > 20.0f) ? z : log1pf(__expf(z));
}

// async cache->LDS 16B copy (per lane), ASYNCcnt-tracked; falls back to
// a direct float4 round-trip through VGPRs if the builtin is unavailable.
__device__ __forceinline__ void async_copy_b128(const float* g, float* l) {
#if HAS_ASYNC_LDS
    __builtin_amdgcn_global_load_async_to_lds_b128(
        (gas_v4ip)(void*)g, (las_v4ip)(void*)l, 0, 0);
#else
    *(float4*)l = *(const float4*)g;
#endif
}

__device__ __forceinline__ void wait_async0() {
#if defined(__AMDGCN__)
#if __has_builtin(__builtin_amdgcn_s_wait_asynccnt)
    __builtin_amdgcn_s_wait_asynccnt(0);
#elif HAS_ASYNC_LDS
    asm volatile("s_wait_asynccnt 0x0" ::: "memory");
#endif
#endif
}

// load f32 at p+{0..7,16..23} -> 16 f16 frag elements
__device__ __forceinline__ void ld16_f32_to_h(const float* p, v16h& a) {
    float4 f0 = *(const float4*)(p + 0);
    float4 f1 = *(const float4*)(p + 4);
    float4 f2 = *(const float4*)(p + 16);
    float4 f3 = *(const float4*)(p + 20);
    a[0]=(_Float16)f0.x; a[1]=(_Float16)f0.y; a[2]=(_Float16)f0.z; a[3]=(_Float16)f0.w;
    a[4]=(_Float16)f1.x; a[5]=(_Float16)f1.y; a[6]=(_Float16)f1.z; a[7]=(_Float16)f1.w;
    a[8]=(_Float16)f2.x; a[9]=(_Float16)f2.y; a[10]=(_Float16)f2.z; a[11]=(_Float16)f2.w;
    a[12]=(_Float16)f3.x; a[13]=(_Float16)f3.y; a[14]=(_Float16)f3.z; a[15]=(_Float16)f3.w;
}

// ---------------------------------------------------------------------------
// Kernel 1: one wave per 16-row tile of L.
//   proj tile (16x64) = x_tile(16x512) @ W_xp^T      -> Bp, Cp to ws
//   dt tile   (16x512) = softplus(dt_in(16x32) @ W_dt^T + b_dt) -> ws
// ---------------------------------------------------------------------------
__global__ void __launch_bounds__(32)
k1_proj_wmma(const float* __restrict__ x, const float* __restrict__ W_xp,
             const float* __restrict__ W_dt, const float* __restrict__ b_dt,
             float* __restrict__ ws_dt, float* __restrict__ ws_B,
             float* __restrict__ ws_C) {
    const int lane = threadIdx.x;           // 0..31
    const int t0   = blockIdx.x * 16;
    const int m    = lane & 15;             // row within tile (A), col within tile (B/C/D)
    const int hi   = lane >> 4;             // half-wave selector

    __shared__ _Float16 dtin[16][32];       // dt_in tile, row-major, for A-frag reload

    // ---- phase A: proj = x @ W_xp^T, 4 N-tiles of 16, K loop of 32 ----
    v8f acc[4] = {v8f{}, v8f{}, v8f{}, v8f{}};
    const float* xrow = x + (size_t)(t0 + m) * D_INNER;
    for (int k0 = 0; k0 < D_INNER; k0 += 32) {
        // A frag: 16-bit A 16x32 layout: lane<16 -> K=k0+{0..7,16..23}; lane>=16 -> +8
        v16h a;
        {
            const int kb = k0 + hi * 8;
            float4 f0 = *(const float4*)(xrow + kb);
            float4 f1 = *(const float4*)(xrow + kb + 4);
            float4 f2 = *(const float4*)(xrow + kb + 16);
            float4 f3 = *(const float4*)(xrow + kb + 20);
            a[0]=(_Float16)f0.x; a[1]=(_Float16)f0.y; a[2]=(_Float16)f0.z; a[3]=(_Float16)f0.w;
            a[4]=(_Float16)f1.x; a[5]=(_Float16)f1.y; a[6]=(_Float16)f1.z; a[7]=(_Float16)f1.w;
            a[8]=(_Float16)f2.x; a[9]=(_Float16)f2.y; a[10]=(_Float16)f2.z; a[11]=(_Float16)f2.w;
            a[12]=(_Float16)f3.x; a[13]=(_Float16)f3.y; a[14]=(_Float16)f3.z; a[15]=(_Float16)f3.w;
        }
        const int koff = k0 + hi * 8;       // B frag: lane<16 -> K=k0+{0..7,16..23} ; lane>=16 -> +8
        #pragma unroll
        for (int jt = 0; jt < 4; ++jt) {
            // B[k, n] = W_xp[n, k] (n = jt*16 + m), k = koff+{0..7,16..23}
            v16h b;
            ld16_f32_to_h(W_xp + (size_t)(jt * 16 + m) * D_INNER + koff, b);
            acc[jt] = __builtin_amdgcn_wmma_f32_16x16x32_f16(
                false, a, false, b, (short)0, acc[jt], false, false);
        }
    }

    // ---- scatter Bp (cols 32..47) and Cp (cols 48..63); stash dt_in in LDS ----
    #pragma unroll
    for (int r = 0; r < 8; ++r) {
        const int row = t0 + hi * 8 + r;
        ws_B[(size_t)row * D_STATE + m] = acc[2][r];
        ws_C[(size_t)row * D_STATE + m] = acc[3][r];
        dtin[hi * 8 + r][m]      = (_Float16)acc[0][r];
        dtin[hi * 8 + r][16 + m] = (_Float16)acc[1][r];
    }
    __syncthreads();

    // ---- phase B: dt = softplus(dt_in @ W_dt^T + b_dt), K=32 (single step) ----
    v16h adt;
    {
        const int kb = hi * 8;
        #pragma unroll
        for (int e = 0; e < 16; ++e) {
            const int K = kb + (e < 8 ? e : e + 8);
            adt[e] = dtin[m][K];
        }
    }
    const int koff2 = hi * 8;
    for (int d0 = 0; d0 < D_INNER; d0 += 16) {
        v16h b;
        ld16_f32_to_h(W_dt + (size_t)(d0 + m) * DT_RANK + koff2, b);
        v8f c = {};
        c = __builtin_amdgcn_wmma_f32_16x16x32_f16(
            false, adt, false, b, (short)0, c, false, false);
        const float bias = b_dt[d0 + m];
        #pragma unroll
        for (int r = 0; r < 8; ++r) {
            const int row = t0 + hi * 8 + r;
            ws_dt[(size_t)row * D_INNER + d0 + m] = softplusf_(c[r] + bias);
        }
    }
}

// ---------------------------------------------------------------------------
// Kernel 2: chunk-local stats with zero initial carries.
// grid (NC, D/16), block 256 = 16 d x 16 n.  Tiles staged via async->LDS DMA.
// ---------------------------------------------------------------------------
__global__ void __launch_bounds__(256)
k2_chunk_stats(const float* __restrict__ x, const float* __restrict__ ws_dt,
               const float* __restrict__ ws_B, const float* __restrict__ A_log,
               const float* __restrict__ alpha_p, const float* __restrict__ beta_logit_p,
               float* __restrict__ vlast, float* __restrict__ Atot,
               float* __restrict__ g31, float* __restrict__ hlast) {
    __shared__ float s_dt[CS][16], s_x[CS][16], s_B[CS][16];
    const int tid = threadIdx.x;
    const int c   = blockIdx.x;
    const int d0  = blockIdx.y * 16;
    const int n   = tid & 15;
    const int dl  = tid >> 4;

#if HAS_ASYNC_LDS
    if (tid < 128) {                         // 128 lanes x 16B covers each 2KB tile
        const int tt = tid >> 2;
        const int sg = (tid & 3) * 4;
        const size_t gr = ((size_t)c * CS + tt) * D_INNER + d0 + sg;
        async_copy_b128(ws_dt + gr, &s_dt[tt][sg]);
        async_copy_b128(x + gr, &s_x[tt][sg]);
        // B chunk is fully contiguous: 32*16 floats
        async_copy_b128(ws_B + (size_t)c * CS * D_STATE + tid * 4, &s_B[0][0] + tid * 4);
    }
#else
    for (int i = tid; i < CS * 16; i += 256) {
        const int tt = i >> 4, k = i & 15;
        const size_t grow = (size_t)(c * CS + tt);
        s_dt[tt][k] = ws_dt[grow * D_INNER + d0 + k];
        s_x [tt][k] = x    [grow * D_INNER + d0 + k];
        s_B [tt][k] = ws_B [grow * D_STATE + k];
    }
#endif

    // overlap DMA with per-thread setup
    const float alpha = *alpha_p;
    const float beta  = sigmoidf_(*beta_logit_p);
    const float A_dn  = -__expf(A_log[(size_t)(d0 + dl) * D_STATE + n]);

    wait_async0();
    __syncthreads();

    float v = 0.f, h = 0.f, Ac = 1.f, g = 0.f, bp = beta;
    #pragma unroll 4
    for (int t = 0; t < CS; ++t) {
        const float dtv  = s_dt[t][dl];
        const float bu   = dtv * s_B[t][n] * s_x[t][dl];
        const float abar = __expf(dtv * A_dn);
        v  = beta * v + alpha * bu;
        h  = abar * h + v;
        Ac *= abar;
        g  = abar * g + bp;
        bp *= beta;
    }
    const size_t idx = (size_t)c * (D_INNER * D_STATE) + (size_t)(d0 + dl) * D_STATE + n;
    vlast[idx] = v; Atot[idx] = Ac; g31[idx] = g; hlast[idx] = h;
}

// ---------------------------------------------------------------------------
// Kernel 3: sequential carry chain over 128 chunks, one thread per (d,n).
// ---------------------------------------------------------------------------
__global__ void __launch_bounds__(256)
k3_carries(const float* __restrict__ vlast, const float* __restrict__ Atot,
           const float* __restrict__ g31, const float* __restrict__ hlast,
           const float* __restrict__ beta_logit_p,
           float* __restrict__ v_in, float* __restrict__ h_in) {
    const int j = blockIdx.x * blockDim.x + threadIdx.x;  // (d*16 + n)
    if (j >= D_INNER * D_STATE) return;
    const size_t DN = (size_t)D_INNER * D_STATE;
    const float beta = sigmoidf_(*beta_logit_p);
    float b32 = 1.f;
    #pragma unroll
    for (int i = 0; i < 5; ++i) b32 = (i == 0) ? beta * beta : b32 * b32;  // beta^32
    float vi = 0.f, hi = 0.f;
    for (int c = 0; c < NC; ++c) {
        const size_t idx = (size_t)c * DN + j;
        if (c + 1 < NC) {                     // hide strided L2 latency of next step
            __builtin_prefetch(&hlast[idx + DN], 0, 0);
            __builtin_prefetch(&Atot [idx + DN], 0, 0);
            __builtin_prefetch(&g31  [idx + DN], 0, 0);
            __builtin_prefetch(&vlast[idx + DN], 0, 0);
        }
        v_in[idx] = vi;
        h_in[idx] = hi;
        hi = hlast[idx] + vi * g31[idx] + hi * Atot[idx];
        vi = b32 * vi + vlast[idx];
    }
}

// ---------------------------------------------------------------------------
// Kernel 4: exact recurrence with true carries; reduce over n (16 lanes) and
// write y = sum_n h*C + D*x.  grid (NC, D/16), block 256.
// ---------------------------------------------------------------------------
__global__ void __launch_bounds__(256)
k4_final(const float* __restrict__ x, const float* __restrict__ ws_dt,
         const float* __restrict__ ws_B, const float* __restrict__ ws_C,
         const float* __restrict__ A_log, const float* __restrict__ D_param,
         const float* __restrict__ alpha_p, const float* __restrict__ beta_logit_p,
         const float* __restrict__ v_in, const float* __restrict__ h_in,
         float* __restrict__ out) {
    __shared__ float s_dt[CS][16], s_x[CS][16], s_B[CS][16], s_C[CS][16];
    const int tid = threadIdx.x;
    const int c   = blockIdx.x;
    const int d0  = blockIdx.y * 16;
    const int n   = tid & 15;
    const int dl  = tid >> 4;

#if HAS_ASYNC_LDS
    if (tid < 128) {
        const int tt = tid >> 2;
        const int sg = (tid & 3) * 4;
        const size_t gr = ((size_t)c * CS + tt) * D_INNER + d0 + sg;
        async_copy_b128(ws_dt + gr, &s_dt[tt][sg]);
        async_copy_b128(x + gr, &s_x[tt][sg]);
        async_copy_b128(ws_B + (size_t)c * CS * D_STATE + tid * 4, &s_B[0][0] + tid * 4);
        async_copy_b128(ws_C + (size_t)c * CS * D_STATE + tid * 4, &s_C[0][0] + tid * 4);
    }
#else
    for (int i = tid; i < CS * 16; i += 256) {
        const int tt = i >> 4, k = i & 15;
        const size_t grow = (size_t)(c * CS + tt);
        s_dt[tt][k] = ws_dt[grow * D_INNER + d0 + k];
        s_x [tt][k] = x    [grow * D_INNER + d0 + k];
        s_B [tt][k] = ws_B [grow * D_STATE + k];
        s_C [tt][k] = ws_C [grow * D_STATE + k];
    }
#endif

    const float alpha = *alpha_p;
    const float beta  = sigmoidf_(*beta_logit_p);
    const int   d     = d0 + dl;
    const float A_dn  = -__expf(A_log[(size_t)d * D_STATE + n]);
    const float Dd    = D_param[d];

    const size_t cidx = (size_t)c * (D_INNER * D_STATE) + (size_t)d * D_STATE + n;
    float v = v_in[cidx];
    float h = h_in[cidx];

    wait_async0();
    __syncthreads();

    for (int t = 0; t < CS; ++t) {
        const float dtv  = s_dt[t][dl];
        const float xv   = s_x[t][dl];
        const float bu   = dtv * s_B[t][n] * xv;
        const float abar = __expf(dtv * A_dn);
        v = beta * v + alpha * bu;
        h = abar * h + v;
        float s = h * s_C[t][n];
        // reduce over n: xor masks 1..8 stay inside each 16-lane half of wave32
        s += __shfl_xor(s, 1);
        s += __shfl_xor(s, 2);
        s += __shfl_xor(s, 4);
        s += __shfl_xor(s, 8);
        if (n == 0) out[(size_t)(c * CS + t) * D_INNER + d] = s + Dd * xv;
    }
}

// ---------------------------------------------------------------------------
extern "C" void kernel_launch(void* const* d_in, const int* in_sizes, int n_in,
                              void* d_out, int out_size, void* d_ws, size_t ws_size,
                              hipStream_t stream) {
    (void)in_sizes; (void)n_in; (void)out_size; (void)ws_size;
    const float* x          = (const float*)d_in[0];
    const float* W_xp       = (const float*)d_in[1];
    const float* W_dt       = (const float*)d_in[2];
    const float* b_dt       = (const float*)d_in[3];
    const float* A_log      = (const float*)d_in[4];
    const float* D_param    = (const float*)d_in[5];
    const float* alpha      = (const float*)d_in[6];
    const float* beta_logit = (const float*)d_in[7];
    float* out = (float*)d_out;
    float* ws  = (float*)d_ws;

    // workspace layout (floats)
    const size_t DN   = (size_t)D_INNER * D_STATE;          // 8192
    float* ws_dt    = ws;                                   // L*D
    float* ws_B     = ws_dt   + (size_t)L_SEQ * D_INNER;    // L*N
    float* ws_C     = ws_B    + (size_t)L_SEQ * D_STATE;    // L*N
    float* ws_vlast = ws_C    + (size_t)L_SEQ * D_STATE;    // NC*DN
    float* ws_Atot  = ws_vlast + (size_t)NC * DN;
    float* ws_g31   = ws_Atot  + (size_t)NC * DN;
    float* ws_hlast = ws_g31   + (size_t)NC * DN;
    float* ws_vin   = ws_hlast + (size_t)NC * DN;
    float* ws_hin   = ws_vin   + (size_t)NC * DN;

    k1_proj_wmma<<<L_SEQ / 16, 32, 0, stream>>>(x, W_xp, W_dt, b_dt, ws_dt, ws_B, ws_C);
    k2_chunk_stats<<<dim3(NC, D_INNER / 16), 256, 0, stream>>>(
        x, ws_dt, ws_B, A_log, alpha, beta_logit, ws_vlast, ws_Atot, ws_g31, ws_hlast);
    k3_carries<<<(D_INNER * D_STATE) / 256, 256, 0, stream>>>(
        ws_vlast, ws_Atot, ws_g31, ws_hlast, beta_logit, ws_vin, ws_hin);
    k4_final<<<dim3(NC, D_INNER / 16), 256, 0, stream>>>(
        x, ws_dt, ws_B, ws_C, A_log, D_param, alpha, beta_logit, ws_vin, ws_hin, out);
}
